// SemGCN_80247168959005
// MI455X (gfx1250) — matrix operand order repaired and checked by reading, not compiled, not run on hardware
//
#include <hip/hip_runtime.h>

// ---------------------------------------------------------------------------
// SemGCN forward for MI455X (gfx1250): all matmuls via v_wmma_f32_16x16x32_bf16
// ---------------------------------------------------------------------------

typedef __bf16 bh;
typedef __attribute__((ext_vector_type(8)))  bh    v8bh;
typedef __attribute__((ext_vector_type(16))) bh    v16bh;
typedef __attribute__((ext_vector_type(8)))  float v8f;

#define BB  8
#define SS  2048
#define DD  768
#define HH  4
#define DKD 192   // head dim

// A fragment (16x32 bf16, MxK): lane holds row lane%16; element e maps to
// K = e + 16*(lane/16)  -> two contiguous 8-elem (16B) chunks.
__device__ __forceinline__ v16bh load_a_frag(const bh* __restrict__ p) {
  v8bh lo = *(const v8bh*)(p);
  v8bh hi = *(const v8bh*)(p + 8);
  v16bh r;
#pragma unroll
  for (int i = 0; i < 8; ++i) { r[i] = lo[i]; r[i + 8] = hi[i]; }
  return r;
}

// ---------------------------------------------------------------------------
// Generic bf16 GEMM:  out[M,N] = A[M,K] * B[N,K]^T  (+bias) (/denom) (relu)
// B stored row-major [N][K] so B-fragment loads are one contiguous 32B chunk.
// 8 waves/block; wave computes 16(M) x 64(N); grid = (N/64, M/128, batch).
// ---------------------------------------------------------------------------
__global__ __launch_bounds__(256)
void gemm_bf16_kernel(const bh* __restrict__ A, const bh* __restrict__ Bm,
                      const float* __restrict__ bias, const float* __restrict__ denom,
                      float* __restrict__ outF, bh* __restrict__ outB,
                      int M, int N, int K,
                      long long sA, long long sB, long long sO, int relu)
{
  const int bz    = blockIdx.z;
  const bh* Ab    = A  + (long long)bz * sA;
  const bh* Bb    = Bm + (long long)bz * sB;
  const int lane  = threadIdx.x & 31;
  const int wave  = threadIdx.x >> 5;
  const int lmod  = lane & 15;
  const int lhalf = lane >> 4;
  const int n0    = blockIdx.x * 64;
  const int m0    = blockIdx.y * 128 + wave * 16;
  if (m0 >= M) return;  // wave-uniform

  v8f c0 = {}, c1 = {}, c2 = {}, c3 = {};
  for (int k0 = 0; k0 < K; k0 += 32) {
    v16bh a = load_a_frag(Ab + (long long)(m0 + lmod) * K + k0 + 16 * lhalf);
    const bh* bp = Bb + (long long)(n0 + lmod) * K + k0 + 16 * lhalf;
    v16bh b0 = *(const v16bh*)(bp);
    v16bh b1 = *(const v16bh*)(bp + (long long)16 * K);
    v16bh b2 = *(const v16bh*)(bp + (long long)32 * K);
    v16bh b3 = *(const v16bh*)(bp + (long long)48 * K);
    c0 = __builtin_amdgcn_wmma_f32_16x16x32_bf16(false, a, false, b0, (short)0, c0, false, false);
    c1 = __builtin_amdgcn_wmma_f32_16x16x32_bf16(false, a, false, b1, (short)0, c1, false, false);
    c2 = __builtin_amdgcn_wmma_f32_16x16x32_bf16(false, a, false, b2, (short)0, c2, false, false);
    c3 = __builtin_amdgcn_wmma_f32_16x16x32_bf16(false, a, false, b3, (short)0, c3, false, false);
  }

  float* oF = outF ? outF + (long long)bz * sO : nullptr;
  bh*    oB = outB ? outB + (long long)bz * sO : nullptr;
#pragma unroll
  for (int j = 0; j < 4; ++j) {
    v8f c = (j == 0) ? c0 : (j == 1) ? c1 : (j == 2) ? c2 : c3;
    const int n  = n0 + 16 * j + lmod;
    const float bs = bias ? bias[n] : 0.0f;
#pragma unroll
    for (int r = 0; r < 8; ++r) {
      const int row = m0 + r + 8 * lhalf;
      float v = c[r] + bs;
      if (denom) v = v / denom[row];
      if (relu)  v = fmaxf(v, 0.0f);
      const long long idx = (long long)row * N + n;
      if (oF) oF[idx] = v;
      if (oB) oB[idx] = (bh)v;
    }
  }
}

// ---------------------------------------------------------------------------
// Fused attention -> adjacency.  Block = one (batch, 16-row q-tile); 8 waves
// stream disjoint 16-col key tiles.  Two-pass softmax (never materializes the
// [b,h,s,s] score tensor).  Writes adj (f32) with diag=1 and row/key masks.
// ---------------------------------------------------------------------------
__global__ __launch_bounds__(256)
void attn_adj_kernel(const bh* __restrict__ qb, const bh* __restrict__ kb,
                     const long long* __restrict__ enc, float* __restrict__ adj)
{
  const int b     = blockIdx.y;
  const int q0    = blockIdx.x * 16;
  const int lane  = threadIdx.x & 31;
  const int wave  = threadIdx.x >> 5;
  const int lmod  = lane & 15;
  const int lhalf = lane >> 4;
  const float scale = 0.07216878364870323f;  // 1/sqrt(192)

  __shared__ float red_m[8][16];
  __shared__ float red_l[8][16];
  __shared__ float mtot[HH][16];
  __shared__ float ltot[HH][16];

  const bh* qbase = qb + (size_t)b * SS * DD;
  const bh* kbase = kb + (size_t)b * SS * DD;
  const long long* encb = enc + (size_t)b * SS;
  float* adjo = adj + (size_t)b * SS * SS;

  // ---- pass 1: per-head row max and sum-of-exp ----
  for (int h = 0; h < HH; ++h) {
    float m[8], l[8];
#pragma unroll
    for (int r = 0; r < 8; ++r) { m[r] = -3.0e38f; l[r] = 0.0f; }
    for (int t = wave; t < SS / 16; t += 8) {
      const int n0 = t * 16;
      v8f acc = {};
#pragma unroll
      for (int c = 0; c < DKD / 32; ++c) {
        const int k0 = h * DKD + c * 32;
        v16bh a  = load_a_frag(qbase + (size_t)(q0 + lmod) * DD + k0 + 16 * lhalf);
        v16bh bb = *(const v16bh*)(kbase + (size_t)(n0 + lmod) * DD + k0 + 16 * lhalf);
        acc = __builtin_amdgcn_wmma_f32_16x16x32_bf16(false, a, false, bb, (short)0, acc, false, false);
      }
      const bool km = (encb[n0 + lmod] != 0);
#pragma unroll
      for (int r = 0; r < 8; ++r) {
        float s  = km ? acc[r] * scale : -1.0e9f;
        float tm = s;
        tm = fmaxf(tm, __shfl_xor(tm, 1));
        tm = fmaxf(tm, __shfl_xor(tm, 2));
        tm = fmaxf(tm, __shfl_xor(tm, 4));
        tm = fmaxf(tm, __shfl_xor(tm, 8));
        const float mn = fmaxf(m[r], tm);
        float e = __expf(s - mn);
        e += __shfl_xor(e, 1);
        e += __shfl_xor(e, 2);
        e += __shfl_xor(e, 4);
        e += __shfl_xor(e, 8);
        l[r] = l[r] * __expf(m[r] - mn) + e;
        m[r] = mn;
      }
    }
    if (lmod == 0) {
#pragma unroll
      for (int r = 0; r < 8; ++r) {
        red_m[wave][r + 8 * lhalf] = m[r];
        red_l[wave][r + 8 * lhalf] = l[r];
      }
    }
    __syncthreads();
    if (threadIdx.x < 16) {
      float Mx = -3.0e38f;
      for (int w2 = 0; w2 < 8; ++w2) Mx = fmaxf(Mx, red_m[w2][threadIdx.x]);
      float Ls = 0.0f;
      for (int w2 = 0; w2 < 8; ++w2) Ls += red_l[w2][threadIdx.x] * __expf(red_m[w2][threadIdx.x] - Mx);
      mtot[h][threadIdx.x] = Mx;
      ltot[h][threadIdx.x] = Ls;
    }
    __syncthreads();
  }

  // ---- pass 2: recompute scores, mean heads, diag/masks, write adj ----
  for (int t = wave; t < SS / 16; t += 8) {
    const int n0 = t * 16;
    const bool km = (encb[n0 + lmod] != 0);
    float ov[8];
#pragma unroll
    for (int r = 0; r < 8; ++r) ov[r] = 0.0f;
    for (int h = 0; h < HH; ++h) {
      v8f acc = {};
#pragma unroll
      for (int c = 0; c < DKD / 32; ++c) {
        const int k0 = h * DKD + c * 32;
        v16bh a  = load_a_frag(qbase + (size_t)(q0 + lmod) * DD + k0 + 16 * lhalf);
        v16bh bb = *(const v16bh*)(kbase + (size_t)(n0 + lmod) * DD + k0 + 16 * lhalf);
        acc = __builtin_amdgcn_wmma_f32_16x16x32_bf16(false, a, false, bb, (short)0, acc, false, false);
      }
#pragma unroll
      for (int r = 0; r < 8; ++r) {
        const int row = r + 8 * lhalf;
        const float s = km ? acc[r] * scale : -1.0e9f;
        ov[r] += __expf(s - mtot[h][row]) * (0.25f / ltot[h][row]);
      }
    }
#pragma unroll
    for (int r = 0; r < 8; ++r) {
      const int row = r + 8 * lhalf;
      const int qg = q0 + row;
      const int ng = n0 + lmod;
      float v = (qg == ng) ? 1.0f : ov[r];      // zero diag, then diag = 1
      if (encb[qg] == 0) v = 0.0f;              // row mask
      adjo[(size_t)qg * SS + ng] = v;
    }
  }
}

// ---------------------------------------------------------------------------
// Small helper kernels
// ---------------------------------------------------------------------------
__global__ __launch_bounds__(256)
void cvt_f32_bf16_kernel(const float* __restrict__ in, bh* __restrict__ out, size_t n) {
  size_t i = (size_t)blockIdx.x * blockDim.x + threadIdx.x;
  if (i < n) out[i] = (bh)in[i];
}

// inputs f32 [b][s][d] -> xb bf16 [b][s][d] and xT bf16 [b][d][s]
__global__ __launch_bounds__(256)
void cvt_x_kernel(const float* __restrict__ in, bh* __restrict__ xb, bh* __restrict__ xT) {
  size_t i = (size_t)blockIdx.x * blockDim.x + threadIdx.x;
  if (i >= (size_t)BB * SS * DD) return;
  const int d = (int)(i % DD);
  const size_t bs = i / DD;
  const int s = (int)(bs % SS);
  const int b = (int)(bs / SS);
  const float v = in[i];
  xb[i] = (bh)v;
  xT[((size_t)b * DD + d) * SS + s] = (bh)v;
}

// bf16 [b][s][d] -> bf16 [b][d][s]
__global__ __launch_bounds__(256)
void transpose_bh_kernel(const bh* __restrict__ in, bh* __restrict__ out) {
  size_t i = (size_t)blockIdx.x * blockDim.x + threadIdx.x;
  if (i >= (size_t)BB * SS * DD) return;
  const int d = (int)(i % DD);
  const size_t bs = i / DD;
  const int s = (int)(bs % SS);
  const int b = (int)(bs / SS);
  out[((size_t)b * DD + d) * SS + s] = in[i];
}

// denom[row] = sum_k adj[row][k] + 1 ; one block per (b*S+s) row
__global__ __launch_bounds__(256)
void denom_kernel(const float* __restrict__ adj, float* __restrict__ denom) {
  __shared__ float sm[256];
  const float* row = adj + (size_t)blockIdx.x * SS;
  float s = 0.0f;
  for (int i = threadIdx.x; i < SS; i += 256) s += row[i];
  sm[threadIdx.x] = s;
  __syncthreads();
  for (int off = 128; off > 0; off >>= 1) {
    if ((int)threadIdx.x < off) sm[threadIdx.x] += sm[threadIdx.x + off];
    __syncthreads();
  }
  if (threadIdx.x == 0) denom[blockIdx.x] = sm[0] + 1.0f;
}

// ---------------------------------------------------------------------------
// Host launcher
// ---------------------------------------------------------------------------
extern "C" void kernel_launch(void* const* d_in, const int* in_sizes, int n_in,
                              void* d_out, int out_size, void* d_ws, size_t ws_size,
                              hipStream_t stream) {
  (void)in_sizes; (void)n_in; (void)out_size; (void)ws_size;
  const float*     inputs = (const float*)d_in[0];
  const long long* enc    = (const long long*)d_in[1];   // int64 encoding
  // d_in[2] = seq_lens (== S), unused
  const float* Wq = (const float*)d_in[3];
  const float* bq = (const float*)d_in[4];
  const float* Wk = (const float*)d_in[5];
  const float* bk = (const float*)d_in[6];
  const float* W0 = (const float*)d_in[7];
  const float* b0 = (const float*)d_in[8];
  const float* W1 = (const float*)d_in[9];
  const float* b1 = (const float*)d_in[10];

  const size_t nBSD = (size_t)BB * SS * DD;   // 12,582,912
  const size_t nBSS = (size_t)BB * SS * SS;   // 33,554,432
  const size_t nW   = (size_t)DD * DD;

  // workspace carve (~173 MB), buffers reused across stages
  char* w = (char*)d_ws;
  bh* buf0 = (bh*)w; w += 2 * nBSD;   // xb, later Ax1/Ax2
  bh* buf1 = (bh*)w; w += 2 * nBSD;   // xT, later o1T
  bh* buf2 = (bh*)w; w += 2 * nBSD;   // qb, later o1
  bh* buf3 = (bh*)w; w += 2 * nBSD;   // kb
  bh* adjb = (bh*)w; w += 2 * nBSS;   // adj in bf16
  bh* wqb  = (bh*)w; w += 2 * nW;
  bh* wkb  = (bh*)w; w += 2 * nW;
  bh* w0b  = (bh*)w; w += 2 * nW;
  bh* w1b  = (bh*)w; w += 2 * nW;
  float* denom = (float*)w; w += sizeof(float) * (size_t)BB * SS;

  float* outF = (float*)d_out;        // out: [b][s][d] f32
  float* adjF = outF + nBSD;          // adj: [b][s][s] f32

  // 1) precision conversion
  {
    const int blk = (int)((nW + 255) / 256);
    cvt_f32_bf16_kernel<<<blk, 256, 0, stream>>>(Wq, wqb, nW);
    cvt_f32_bf16_kernel<<<blk, 256, 0, stream>>>(Wk, wkb, nW);
    cvt_f32_bf16_kernel<<<blk, 256, 0, stream>>>(W0, w0b, nW);
    cvt_f32_bf16_kernel<<<blk, 256, 0, stream>>>(W1, w1b, nW);
  }
  cvt_x_kernel<<<(int)((nBSD + 255) / 256), 256, 0, stream>>>(inputs, buf0, buf1);

  // 2) q,k projections: [16384,768] = x @ W^T + b  (bf16 out)
  gemm_bf16_kernel<<<dim3(DD / 64, (BB * SS) / 128, 1), 256, 0, stream>>>(
      buf0, wqb, bq, nullptr, nullptr, buf2, BB * SS, DD, DD, 0, 0, 0, 0);
  gemm_bf16_kernel<<<dim3(DD / 64, (BB * SS) / 128, 1), 256, 0, stream>>>(
      buf0, wkb, bk, nullptr, nullptr, buf3, BB * SS, DD, DD, 0, 0, 0, 0);

  // 3) fused attention -> adj (writes straight into d_out's adj slice)
  attn_adj_kernel<<<dim3(SS / 16, BB), 256, 0, stream>>>(buf2, buf3, enc, adjF);

  // 4) denom = rowsum(adj)+1 ; bf16 copy of adj for the GCN matmuls
  denom_kernel<<<BB * SS, 256, 0, stream>>>(adjF, denom);
  cvt_f32_bf16_kernel<<<(int)((nBSS + 255) / 256), 256, 0, stream>>>(adjF, adjb, nBSS);

  // 5) GCN layer 1: Ax1 = adj @ x  (batched; B-operand = xT, [d][s] per batch)
  gemm_bf16_kernel<<<dim3(DD / 64, SS / 128, BB), 256, 0, stream>>>(
      adjb, buf1, nullptr, nullptr, nullptr, buf0, SS, DD, SS,
      (long long)SS * SS, (long long)DD * SS, (long long)SS * DD, 0);
  //    o1 = relu((Ax1 @ W0^T + b0) / denom)  (bf16 out)
  gemm_bf16_kernel<<<dim3(DD / 64, (BB * SS) / 128, 1), 256, 0, stream>>>(
      buf0, w0b, b0, denom, nullptr, buf2, BB * SS, DD, DD, 0, 0, 0, 1);
  transpose_bh_kernel<<<(int)((nBSD + 255) / 256), 256, 0, stream>>>(buf2, buf1);

  // 6) GCN layer 2: Ax2 = adj @ o1 ; out = relu((Ax2 @ W1^T + b1)/denom) (f32)
  gemm_bf16_kernel<<<dim3(DD / 64, SS / 128, BB), 256, 0, stream>>>(
      adjb, buf1, nullptr, nullptr, nullptr, buf0, SS, DD, SS,
      (long long)SS * SS, (long long)DD * SS, (long long)SS * DD, 0);
  gemm_bf16_kernel<<<dim3(DD / 64, (BB * SS) / 128, 1), 256, 0, stream>>>(
      buf0, w1b, b1, denom, outF, nullptr, BB * SS, DD, DD, 0, 0, 0, 1);
}
